// GFA_1176821039818
// MI455X (gfx1250) — matrix operand order repaired and checked by reading, not compiled, not run on hardware
//
#include <hip/hip_runtime.h>
#include <cstdint>
#include <cstddef>

// Problem constants (B,C,H,W = 4,64,64,64; CI = 16; P = H*W = 4096)
#define NB   4
#define NC   64
#define NCI  16
#define P_PIX 4096
#define KB_KEYS 64          // keys processed per attention loop iteration

typedef __attribute__((ext_vector_type(16))) __bf16 bf16x16;
typedef __attribute__((ext_vector_type(8)))  __bf16 bf16x8;
typedef __attribute__((ext_vector_type(8)))  float  f32x8;
typedef __attribute__((ext_vector_type(4)))  unsigned int u32x4;
typedef __attribute__((ext_vector_type(8)))  int i32x8;
typedef __attribute__((ext_vector_type(4)))  int i32x4;

#define CAT16 0,1,2,3,4,5,6,7,8,9,10,11,12,13,14,15

#if __has_builtin(__builtin_amdgcn_tensor_load_to_lds)
#define USE_TDM 1
#else
#define USE_TDM 0
#endif

// Exchange data between lane l and lane l^16 (the two 16-lane halves).
__device__ __forceinline__ float swap16(float x) {
#if __has_builtin(__builtin_amdgcn_permlanex16)
  return __int_as_float(__builtin_amdgcn_permlanex16(
      __float_as_int(x), __float_as_int(x), 0x76543210, 0xFEDCBA98,
      false, false));
#else
  return __int_as_float(__builtin_amdgcn_ds_swizzle(__float_as_int(x), 0x401f));
#endif
}

// Raw v_exp_f32 (2^x). Flushes sub-2^-126 results to zero -- fine for softmax.
__device__ __forceinline__ float xexp2(float x) {
#if __has_builtin(__builtin_amdgcn_exp2f)
  return __builtin_amdgcn_exp2f(x);
#else
  return exp2f(x);
#endif
}

#if USE_TDM
__device__ __forceinline__ void wait_tensor0() {
#if __has_builtin(__builtin_amdgcn_s_wait_tensorcnt)
  __builtin_amdgcn_s_wait_tensorcnt(0);
#else
  asm volatile("s_wait_tensorcnt 0x0" ::: "memory");
#endif
}

// Issue one TDM 2D tile load: tile_d1 rows x tile_d0 elements (2-byte data),
// row stride stride0 elements, global tile start gptr, dest LDS byte address.
// D# bit layout per CDNA5 ISA 8.3-8.4 (count=1, type=2=image, data_size=1,
// no multicast / padding / iteration / atomic-barrier).
__device__ __forceinline__ void tdm_load_2d(
    unsigned lds_byte_addr, const void* gptr,
    int tile_d0, int tile_d1, int tensor_d0, int tensor_d1,
    long long stride0_elems)
{
  const unsigned long long ga = (unsigned long long)(uintptr_t)gptr;
  u32x4 g0;
  g0[0] = 1u;                                     // count=1 (valid user D#)
  g0[1] = lds_byte_addr;                          // LDS dest (bytes)
  g0[2] = (unsigned)(ga & 0xFFFFFFFFu);           // global_addr[31:0]
  g0[3] = (unsigned)((ga >> 32) & 0x01FFFFFFu) | 0x80000000u;  // [56:32]|type=2
  i32x8 g1;
  g1[0] = 1 << 16;                                // wg_mask=0, data_size=1 (2B)
  g1[1] = (tensor_d0 & 0xFFFF) << 16;             // abar=0 | tensor_dim0 lo
  g1[2] = ((tensor_d0 >> 16) & 0xFFFF) | ((tensor_d1 & 0xFFFF) << 16);
  g1[3] = ((tensor_d1 >> 16) & 0xFFFF) | ((tile_d0 & 0xFFFF) << 16);
  g1[4] = (tile_d1 & 0xFFFF);                     // tile_dim1 | tile_dim2=0
  g1[5] = (int)(stride0_elems & 0xFFFFFFFFll);    // tensor_dim0_stride lo
  g1[6] = (int)((stride0_elems >> 32) & 0xFFFFll);// stride hi | dim1_stride=0
  g1[7] = 0;
  const i32x4 gz = {0, 0, 0, 0};
#if __has_include(<hip/amd_detail/amd_gfx1250_TDM.h>)
  const i32x8 gz8 = {0, 0, 0, 0, 0, 0, 0, 0};     // therock 6-arg form
  __builtin_amdgcn_tensor_load_to_lds(g0, g1, gz, gz, gz8, 0);
#else
  __builtin_amdgcn_tensor_load_to_lds(g0, g1, gz, gz, 0);   // ROCm 5-arg form
#endif
}
#endif  // USE_TDM

// ---------------------------------------------------------------------------
// Kernel 1: 1x1-conv projections -> bf16 buffers for the WMMA attention.
//   bufQ[b][i][d]  = log2(e) * ((wk @ x)[d,i] + bk[d])  (attention queries)
//   bufK[b][p][d]  = (wq @ x)[d,p] + bq[d]              (attention keys)
//   bufV[b][c][p]  = (wv @ x)[c,p] + bv[c]              (values)
// ---------------------------------------------------------------------------
__global__ __launch_bounds__(256) void proj_kernel(
    const float* __restrict__ ftr,
    const float* __restrict__ wq, const float* __restrict__ bq,
    const float* __restrict__ wk, const float* __restrict__ bk,
    const float* __restrict__ wv, const float* __restrict__ bv,
    __bf16* __restrict__ bufQ, __bf16* __restrict__ bufK,
    __bf16* __restrict__ bufV)
{
  __shared__ __align__(16) float swq[NCI*NC];
  __shared__ __align__(16) float swk[NCI*NC];
  __shared__ __align__(16) float swv[NC*NC];
  __shared__ float sbq[NCI], sbk[NCI], sbv[NC];

  const int t = threadIdx.x;
  for (int i = t; i < NCI*NC; i += 256) { swq[i] = wq[i]; swk[i] = wk[i]; }
  for (int i = t; i < NC*NC;  i += 256) { swv[i] = wv[i]; }
  if (t < NCI) { sbq[t] = bq[t]; sbk[t] = bk[t]; }
  if (t < NC)  { sbv[t] = bv[t]; }
  __syncthreads();

  const int pix = blockIdx.x * 256 + t;
  const int b = pix >> 12;            // /4096
  const int p = pix & (P_PIX - 1);

  float x[NC];
  #pragma unroll
  for (int c = 0; c < NC; ++c)
    x[c] = ftr[((size_t)(b*NC + c))*P_PIX + p];

  const float LOG2E = 1.44269504088896f;
  bf16x16 outK, outQ;
  #pragma unroll
  for (int o = 0; o < NCI; ++o) {
    float aq = sbq[o], ak = sbk[o];
    #pragma unroll
    for (int c = 0; c < NC; ++c) {
      aq = fmaf(swq[o*NC + c], x[c], aq);
      ak = fmaf(swk[o*NC + c], x[c], ak);
    }
    outK[o] = (__bf16)aq;             // q-projection = attention keys
    outQ[o] = (__bf16)(ak * LOG2E);   // k-projection = attention queries
  }
  *(bf16x16*)(bufK + ((size_t)b*P_PIX + p)*NCI) = outK;
  *(bf16x16*)(bufQ + ((size_t)b*P_PIX + p)*NCI) = outQ;

  #pragma unroll 8
  for (int o = 0; o < NC; ++o) {
    float av = sbv[o];
    #pragma unroll
    for (int c = 0; c < NC; ++c) av = fmaf(swv[o*NC + c], x[c], av);
    bufV[((size_t)(b*NC + o))*P_PIX + p] = (__bf16)av;
  }
}

// ---------------------------------------------------------------------------
// Kernel 2: fused flash attention. 256 blocks (4 batches x 64 query tiles of
// 64), 128 threads = 4 waves; each wave owns a 16-query tile and all 64
// channels. 64 keys/iter: 4 WMMAs S^T, one base-2 online-softmax update, then
// 2x (8 LDS loads issued early; exp2/permlanex16/pack; 4 WMMAs out^T += V^T P^T).
// K/V blocks double-buffered in LDS; staged by the Tensor Data Mover (one
// tensor_load_to_lds per block per buffer, issued by wave 0, TENSORcnt +
// barrier hand-off) with a register-staging fallback if TDM is unavailable.
// ---------------------------------------------------------------------------
__global__ __launch_bounds__(128) void attn_kernel(
    const float* __restrict__ ftr, const float* __restrict__ delta,
    const __bf16* __restrict__ bufQ, const __bf16* __restrict__ bufK,
    const __bf16* __restrict__ bufV,
    float* __restrict__ pre)
{
  __shared__ __align__(16) __bf16 ldsK[2][KB_KEYS * NCI];  // [key][d]  2x2 KB
  __shared__ __align__(16) __bf16 ldsV[2][NC * KB_KEYS];   // [c][key]  2x8 KB

  const int t    = threadIdx.x;
  const int wave = t >> 5;
  const int lane = t & 31;
  const int half = lane >> 4;     // which 16-lane group
  const int n    = lane & 15;     // query column (N) / A-matrix row (M)
  const int b    = blockIdx.x >> 6;
  const int i0   = (blockIdx.x & 63) * 64 + wave * 16;

#if USE_TDM
  // TDM staging: one 2D-tile descriptor per block.
  //   K block: contiguous 64 keys x 16 d = one 1024-element row.
  //   V block: 64 rows (c) x 64 elements (keys), row stride P.
  const __bf16* gKbase = bufK + ((size_t)b*P_PIX)*NCI;
  const __bf16* gVbase = bufV + ((size_t)(b*NC))*P_PIX;
  unsigned ldsKaddr[2], ldsVaddr[2];
  #pragma unroll
  for (int u = 0; u < 2; ++u) {
    ldsKaddr[u] = (unsigned)(uintptr_t)(void*)ldsK[u];
    ldsVaddr[u] = (unsigned)(uintptr_t)(void*)ldsV[u];
  }
#else
  // Register staging: K = 128 chunks of 16B (one/thread), V = 512 (four/thread)
  const __bf16* gK = bufK + ((size_t)b*P_PIX + (t >> 1))*NCI + (t & 1)*8;
  const int sK = (t >> 1)*NCI + (t & 1)*8;
  const __bf16* gV[4];
  int sV[4];
  #pragma unroll
  for (int r = 0; r < 4; ++r) {
    const int j = t + 128*r;
    gV[r] = bufV + ((size_t)(b*NC + (j >> 3)))*P_PIX + (j & 7)*8;
    sV[r] = (j >> 3)*KB_KEYS + (j & 7)*8;
  }
#endif

  // B-operand of S^T WMMA: Q^T [d(32, zero-padded) x 16 queries].
  const bf16x8 q8 = *(const bf16x8*)(bufQ + ((size_t)b*P_PIX + i0 + n)*NCI + half*8);
  const bf16x8 qz8 = {};
  bf16x16 bQ = __builtin_shufflevector(q8, qz8, CAT16);

  f32x8 acc[4] = {};                 // out^T tiles for c-tiles 0..3
  float m_run = -1e30f, l_run = 0.f;
  const float dlt = delta[0];

#if USE_TDM
  // Prologue: DMA block 0 into buffer 0.
  if (wave == 0) {
    tdm_load_2d(ldsKaddr[0], gKbase, KB_KEYS*NCI, 1, P_PIX*NCI, 1, P_PIX*NCI);
    tdm_load_2d(ldsVaddr[0], gVbase, KB_KEYS, NC, P_PIX, NC, P_PIX);
  }
#else
  u32x4 rK = *(const u32x4*)gK;
  u32x4 rV[4];
  #pragma unroll
  for (int r = 0; r < 4; ++r) rV[r] = *(const u32x4*)gV[r];
#endif

  for (int kb = 0; kb < P_PIX; kb += KB_KEYS) {
    const int buf = (kb >> 6) & 1;
    const int kn = kb + KB_KEYS;
#if USE_TDM
    if (wave == 0) wait_tensor0();     // our DMA for LDS[buf] has landed
    __syncthreads();                   // visible to all; prior reads done
    if (kn < P_PIX && wave == 0) {     // DMA next block into the other buffer
      tdm_load_2d(ldsKaddr[buf ^ 1], gKbase + (size_t)kn*NCI,
                  KB_KEYS*NCI, 1, P_PIX*NCI, 1, P_PIX*NCI);
      tdm_load_2d(ldsVaddr[buf ^ 1], gVbase + kn,
                  KB_KEYS, NC, P_PIX, NC, P_PIX);
    }
#else
    *(u32x4*)(ldsK[buf] + sK) = rK;
    #pragma unroll
    for (int r = 0; r < 4; ++r) *(u32x4*)(ldsV[buf] + sV[r]) = rV[r];
    __syncthreads();
    if (kn < P_PIX) {
      rK = *(const u32x4*)(gK + (size_t)kn*NCI);
      #pragma unroll
      for (int r = 0; r < 4; ++r) rV[r] = *(const u32x4*)(gV[r] + kn);
    }
#endif
    const __bf16* lK = ldsK[buf];
    const __bf16* lV = ldsV[buf];

    // ---- S^T tiles: batch all 4 A-tile loads, then 4 WMMAs ----
    const bf16x8 zero8 = {};
    const f32x8 z = {};
    bf16x8 rk[4];
    #pragma unroll
    for (int s = 0; s < 4; ++s)
      rk[s] = *(const bf16x8*)(lK + (n + 16*s)*NCI + half*8);
    f32x8 st[4];
    #pragma unroll
    for (int s = 0; s < 4; ++s) {
      bf16x16 aK = __builtin_shufflevector(rk[s], zero8, CAT16);
      st[s] = __builtin_amdgcn_wmma_f32_16x16x32_bf16(false, aK, false, bQ,
                                                      (short)0, z, false, false);
    }

    // ---- one online-softmax update for all 64 keys (base-2 domain) ----
    float mx = st[0][0];
    #pragma unroll
    for (int s = 0; s < 4; ++s)
      #pragma unroll
      for (int j = 0; j < 8; ++j) mx = fmaxf(mx, st[s][j]);
    mx = fmaxf(mx, swap16(mx));                 // combine the two half-waves
    const float m_new = fmaxf(m_run, mx);
    const float corr  = xexp2(m_run - m_new);
    m_run = m_new;
    #pragma unroll
    for (int tt = 0; tt < 4; ++tt)
      #pragma unroll
      for (int j = 0; j < 8; ++j) acc[tt][j] *= corr;

    // ---- two 32-key chunks: V-tile LDS loads first (latency hides under the
    // exp2/swap/pack VALU work), then B operand, then 4 PV WMMAs.
    // B-operand layout (K=32 keys, N=16 queries):
    //  lanes 0-15  need keys 0..15  = {own e0 (M=0-7), swapped e0 (M=8-15)}
    //  lanes 16-31 need keys 16..31 = {swapped e1 (M=0-7), own e1 (M=8-15)}
    float rs = 0.f;
    #pragma unroll
    for (int hc = 0; hc < 2; ++hc) {
      const int ko = hc*32;
      bf16x8 va[8];
      #pragma unroll
      for (int tt = 0; tt < 4; ++tt) {
        const int c = tt*16 + n;               // A-matrix row this lane feeds
        va[2*tt    ] = *(const bf16x8*)(lV + c*KB_KEYS + ko + half*16);
        va[2*tt + 1] = *(const bf16x8*)(lV + c*KB_KEYS + ko + half*16 + 8);
      }
      bf16x16 bP;
      #pragma unroll
      for (int j = 0; j < 8; ++j) {
        const float a0 = xexp2(st[2*hc    ][j] - m_new);
        const float a1 = xexp2(st[2*hc + 1][j] - m_new);
        const float b0 = swap16(a0);
        const float b1 = swap16(a1);
        rs += (a0 + a1) + (b0 + b1);
        bP[j]     = (__bf16)(half ? b1 : a0);
        bP[8 + j] = (__bf16)(half ? a1 : b0);
      }
      #pragma unroll
      for (int tt = 0; tt < 4; ++tt) {
        bf16x16 aV = __builtin_shufflevector(va[2*tt], va[2*tt + 1], CAT16);
        acc[tt] = __builtin_amdgcn_wmma_f32_16x16x32_bf16(false, aV, false, bP,
                                                          (short)0, acc[tt],
                                                          false, false);
      }
    }
    l_run = l_run * corr + rs;
  }

  // ---- finalize: pre = delta * (acc / l) + ftr ----
  const float inv = 1.0f / l_run;
  #pragma unroll
  for (int tt = 0; tt < 4; ++tt) {
    #pragma unroll
    for (int j = 0; j < 8; ++j) {
      const int c = tt*16 + j + half*8;        // C/D layout: M = j + 8*half
      const size_t idx = ((size_t)(b*NC + c))*P_PIX + i0 + n;
      pre[idx] = dlt * acc[tt][j] * inv + ftr[idx];
    }
  }
}

// ---------------------------------------------------------------------------
// Kernel 3: spatial avg+max pooling per (b,c) row of pre-gate output.
// ---------------------------------------------------------------------------
__global__ __launch_bounds__(256) void pool_kernel(
    const float* __restrict__ pre, float* __restrict__ favg,
    float* __restrict__ fmax)
{
  __shared__ float rs[256], rm[256];
  const int row = blockIdx.x, t = threadIdx.x;
  const float* src = pre + (size_t)row * P_PIX;
  float s = 0.f, m = -1e30f;
  for (int i = t; i < P_PIX; i += 256) {
    float v = src[i]; s += v; m = fmaxf(m, v);
  }
  rs[t] = s; rm[t] = m;
  __syncthreads();
  for (int o = 128; o > 0; o >>= 1) {
    if (t < o) { rs[t] += rs[t + o]; rm[t] = fmaxf(rm[t], rm[t + o]); }
    __syncthreads();
  }
  if (t == 0) { favg[row] = rs[0] * (1.0f / P_PIX); fmax[row] = rm[0]; }
}

// ---------------------------------------------------------------------------
// Kernel 4: SE gate  sigmoid(relu(favg@W1a^T)@W2a^T + relu(fmax@W1m^T)@W2m^T)
// ---------------------------------------------------------------------------
__global__ __launch_bounds__(64) void gate_kernel(
    const float* __restrict__ favg, const float* __restrict__ fmax,
    const float* __restrict__ w_avg1, const float* __restrict__ w_avg2,
    const float* __restrict__ w_max1, const float* __restrict__ w_max2,
    float* __restrict__ gate)
{
  __shared__ float ha[NCI], hm[NCI];
  const int b = blockIdx.x, t = threadIdx.x;
  if (t < NCI) {
    float sa = 0.f, sm = 0.f;
    for (int c = 0; c < NC; ++c) {
      sa = fmaf(w_avg1[t*NC + c], favg[b*NC + c], sa);
      sm = fmaf(w_max1[t*NC + c], fmax[b*NC + c], sm);
    }
    ha[t] = fmaxf(sa, 0.f);
    hm[t] = fmaxf(sm, 0.f);
  }
  __syncthreads();
  float a = 0.f, m = 0.f;
  for (int h = 0; h < NCI; ++h) {
    a = fmaf(w_avg2[t*NCI + h], ha[h], a);
    m = fmaf(w_max2[t*NCI + h], hm[h], m);
  }
  gate[b*NC + t] = 1.0f / (1.0f + __expf(-(a + m)));
}

// ---------------------------------------------------------------------------
// Kernel 5: apply gate.
// ---------------------------------------------------------------------------
__global__ __launch_bounds__(256) void apply_kernel(
    const float* __restrict__ pre, const float* __restrict__ gate,
    float* __restrict__ out)
{
  const int i = blockIdx.x * 256 + threadIdx.x;
  out[i] = pre[i] * gate[i >> 12];   // i / P  -> (b,c) row
}

// ---------------------------------------------------------------------------
extern "C" void kernel_launch(void* const* d_in, const int* in_sizes, int n_in,
                              void* d_out, int out_size, void* d_ws,
                              size_t ws_size, hipStream_t stream)
{
  const float* ftr    = (const float*)d_in[0];
  const float* wq     = (const float*)d_in[1];
  const float* bq     = (const float*)d_in[2];
  const float* wk     = (const float*)d_in[3];
  const float* bk     = (const float*)d_in[4];
  const float* wv     = (const float*)d_in[5];
  const float* bv     = (const float*)d_in[6];
  const float* delta  = (const float*)d_in[7];
  const float* w_avg1 = (const float*)d_in[8];
  const float* w_avg2 = (const float*)d_in[9];
  const float* w_max1 = (const float*)d_in[10];
  const float* w_max2 = (const float*)d_in[11];

  char* ws = (char*)d_ws;
  __bf16* bufQ = (__bf16*)(ws + 0);                 // 512 KB [B,P,16]
  __bf16* bufK = (__bf16*)(ws + (512 << 10));       // 512 KB [B,P,16]
  __bf16* bufV = (__bf16*)(ws + (1 << 20));         //   2 MB [B,64,P]
  float*  pre  = (float* )(ws + (3 << 20));         //   4 MB [B,64,P]
  float*  favg = (float* )(ws + (7 << 20));
  float*  fmax = (float* )(ws + (7 << 20) + 1024);
  float*  gate = (float* )(ws + (7 << 20) + 2048);

  proj_kernel <<<NB*P_PIX/256, 256, 0, stream>>>(ftr, wq, bq, wk, bk, wv, bv,
                                                 bufQ, bufK, bufV);
  attn_kernel <<<NB*(P_PIX/64), 128, 0, stream>>>(ftr, delta, bufQ, bufK, bufV,
                                                  pre);
  pool_kernel <<<NB*NC, 256, 0, stream>>>(pre, favg, fmax);
  gate_kernel <<<NB, 64, 0, stream>>>(favg, fmax, w_avg1, w_avg2, w_max1,
                                      w_max2, gate);
  apply_kernel<<<NB*NC*P_PIX/256, 256, 0, stream>>>(pre, gate, (float*)d_out);
}
